// MultiDimensionalLSTM_27530740367823
// MI455X (gfx1250) — matrix operand order repaired and checked by reading, not compiled, not run on hardware
//
#include <hip/hip_runtime.h>
#include <hip/hip_bf16.h>
#include <stdint.h>

// ---------------- CDNA5 WMMA types ----------------
typedef __attribute__((ext_vector_type(16))) __bf16 bf16x16;
typedef __attribute__((ext_vector_type(8)))  float  f32x8;

__device__ __forceinline__ f32x8 wmma_bf16(bf16x16 a, bf16x16 b, f32x8 c) {
  // v_wmma_f32_16x16x32_bf16  (neg_a, A, neg_b, B, c_mod, C, reuse_a, reuse_b)
  return __builtin_amdgcn_wmma_f32_16x16x32_bf16(false, a, false, b, (short)0, c,
                                                 false, false);
}

// Problem constants
#define T_STEPS 1024
#define BATCH   32
#define RR      128
#define FF      512
#define N5      640
#define NBT     40     // 640/16 N tiles
#define KX      512
#define KH      256

// ---- workspace layout (bytes), total ~152 MB ----
#define OFF_XS  (size_t)0                       // bf16 xs      [1024][32][512]   33,554,432
#define OFF_WX  (size_t)33554432                // bf16 WxPack  [16*40 frags]        655,360
#define OFF_WH  (OFF_WX + 655360)               // bf16 WhPack  [ 8*40 frags]        327,680
#define OFF_GX  (OFF_WH + 327680)               // f32 gatesX   [1024][32][640]   83,886,080
#define OFF_C   (OFF_GX + 83886080)             // f32 C_all    [1024][32][128]   16,777,216
#define OFF_H   (OFF_C  + 16777216)             // f32 H_all    [1024][32][128]   16,777,216

__device__ __forceinline__ float sigmoidf_(float x) {
  return 1.0f / (1.0f + __expf(-x));
}

// A fragment (16x32 bf16, M x K) from row-major LDS buffer.
// Lanes 0-15: M=lane, K = k0..k0+7 and k0+16..k0+23 with k0 = kb*32
// Lanes 16-31: same M, K shifted by +8.
__device__ __forceinline__ bf16x16 load_frag_a(const __bf16* lds, int stride,
                                               int mb, int kb, int lane) {
  int row = mb * 16 + (lane & 15);
  int k0  = kb * 32 + ((lane >> 4) & 1) * 8;
  const __bf16* p = lds + row * stride + k0;
  bf16x16 a;
  ((uint4*)&a)[0] = *(const uint4*)(p);        // ds_load_b128
  ((uint4*)&a)[1] = *(const uint4*)(p + 16);   // ds_load_b128
  return a;
}

// B fragment (32x16 bf16, K x N) from pre-packed lane-major global buffer.
__device__ __forceinline__ bf16x16 load_frag_b(const __bf16* pack, int frag,
                                               int lane) {
  const __bf16* p = pack + ((size_t)frag * 32 + lane) * 16;
  bf16x16 b;
  ((uint4*)&b)[0] = ((const uint4*)p)[0];      // global_load_b128
  ((uint4*)&b)[1] = ((const uint4*)p)[1];
  return b;
}

// ---------------- kernel 1: reorder + convert x -> xs bf16 ----------------
// xs[t][b][f] = x[b][c][y][xc] with h=t>>5, w=t&31, y=4h+(w>>3),
// xc=(w&7)*16+(f>>5), c=f&31   (faithful to the torch permute/view chain)
__global__ void mdlstm_pack_xs(const float* __restrict__ x,
                               __bf16* __restrict__ xs) {
  size_t idx = (size_t)blockIdx.x * 256 + threadIdx.x;  // 16,777,216 total
  int f  = (int)(idx & 511);
  int b  = (int)((idx >> 9) & 31);
  int t  = (int)(idx >> 14);
  int h  = t >> 5, w = t & 31;
  int y  = 4 * h + (w >> 3);
  int xc = (w & 7) * 16 + (f >> 5);
  int cc = f & 31;
  float v = x[(((size_t)b * 32 + cc) * 128 + y) * 128 + xc];
  xs[idx] = (__bf16)v;
}

// ---------------- kernel 2: pack W into WMMA B-fragment order ----------------
__global__ void mdlstm_pack_w(const float* __restrict__ W,
                              __bf16* __restrict__ wxpack,
                              __bf16* __restrict__ whpack) {
  int gid = blockIdx.x * 256 + threadIdx.x;     // 30720 = (16+8)*40*32
  bool isWh = gid >= 16 * NBT * 32;
  int id   = isWh ? gid - 16 * NBT * 32 : gid;
  int lane = id & 31;
  int frag = id >> 5;                           // kb*40 + nb
  int kb   = frag / NBT, nb = frag % NBT;
  int kbase = (isWh ? 512 : 0) + kb * 32 + ((lane < 16) ? 0 : 8);
  int N     = nb * 16 + (lane & 15);
  __bf16* dst = (isWh ? whpack : wxpack) + ((size_t)frag * 32 + lane) * 16;
#pragma unroll
  for (int e = 0; e < 16; ++e)                  // element e <-> K = kbase + e + (e >= 8 ? 8 : 0)
    dst[e] = (__bf16)W[(size_t)(kbase + e + (e >= 8 ? 8 : 0)) * N5 + N];
}

// ---------------- kernel 3: input projection gatesX = xs @ Wx + bias ----------
// one workgroup per t: [32 x 512] @ [512 x 640]
// 512 threads = 16 waves; wave w: m-tile = w&1, n-tiles = (w>>1)*5 .. +5
// B fragments double-buffered across kb; k-loop fully unrolled so the buffer
// rotation is an SSA rename (no v_mov copies).
#define XS_STRIDE 520   // 512 + 8 bf16 pad (breaks LDS bank alignment)
__global__ void __launch_bounds__(512, 1)
mdlstm_gemm_x(const __bf16* __restrict__ xs,
              const __bf16* __restrict__ wxpack,
              const float* __restrict__ bias,
              float* __restrict__ gatesX) {
  __shared__ __bf16 As[32 * XS_STRIDE];
  const int t = blockIdx.x, tid = threadIdx.x;
  // stage xs[t] (32x512 bf16) into LDS, 16B chunks
  const __bf16* src = xs + (size_t)t * 32 * 512;
  for (int i = tid; i < 32 * 64; i += 512) {
    int row = i >> 6, c8 = i & 63;
    *(uint4*)&As[row * XS_STRIDE + c8 * 8] = ((const uint4*)(src + row * 512))[c8];
  }
  __syncthreads();

  const int wave = tid >> 5, lane = tid & 31;
  const int mt = wave & 1, nb0 = (wave >> 1) * 5;
  f32x8 acc[5];
#pragma unroll
  for (int nt = 0; nt < 5; ++nt) {
    float bv = bias[(nb0 + nt) * 16 + (lane & 15)];
    f32x8 cinit = {bv, bv, bv, bv, bv, bv, bv, bv};
    acc[nt] = cinit;
  }
  // software pipeline: preload B frags and A frag for kb=0
  bf16x16 bcur[5], bnxt[5];
#pragma unroll
  for (int nt = 0; nt < 5; ++nt) bcur[nt] = load_frag_b(wxpack, nb0 + nt, lane);
  bf16x16 a = load_frag_a(As, XS_STRIDE, mt, 0, lane);
#pragma unroll
  for (int kb = 0; kb < 16; ++kb) {
    if (kb + 1 < 16) {
#pragma unroll
      for (int nt = 0; nt < 5; ++nt)
        bnxt[nt] = load_frag_b(wxpack, (kb + 1) * NBT + nb0 + nt, lane);
    }
    bf16x16 anxt = (kb + 1 < 16) ? load_frag_a(As, XS_STRIDE, mt, kb + 1, lane) : a;
#pragma unroll
    for (int nt = 0; nt < 5; ++nt) acc[nt] = wmma_bf16(a, bcur[nt], acc[nt]);
    a = anxt;
#pragma unroll
    for (int nt = 0; nt < 5; ++nt) bcur[nt] = bnxt[nt];
  }
  // write out (D layout: VGPR r -> M = mt*16 + (lane>=16?8:0) + r, N = nb*16+(lane&15))
  float* dst = gatesX + (size_t)t * BATCH * N5;
  const int M = mt * 16 + ((lane >> 4) & 1) * 8;
#pragma unroll
  for (int nt = 0; nt < 5; ++nt) {
    int N = (nb0 + nt) * 16 + (lane & 15);
#pragma unroll
    for (int r = 0; r < 8; ++r) dst[(size_t)(M + r) * N5 + N] = acc[nt][r];
  }
}

// ---------------- kernel 4: one anti-diagonal of the 2D-LSTM wavefront -------
// block = one cell (r,c), t = r*32+c. [32 x 256] @ [256 x 640] + gatesX[t].
// 512 threads = 16 waves, same (mt, 5-n-tile) mapping; B double-buffered,
// k-loop fully unrolled.
#define AS_STRIDE 264   // 256 + 8 bf16 pad
#define GL_STRIDE 648   // 640 + 8 f32 pad
__global__ void __launch_bounds__(512, 1)
mdlstm_scan_diag(const __bf16* __restrict__ whpack,
                 const float* __restrict__ gatesX,
                 float* __restrict__ C_all,
                 float* __restrict__ H_all,
                 float* __restrict__ out, int d) {
  extern __shared__ char smem[];
  __bf16* As = (__bf16*)smem;                        // [32][264] bf16
  float*  gl = (float*)(smem + 32 * AS_STRIDE * 2);  // [32][648] f32

  const int r0 = (d < 32) ? 0 : (d - 31);
  const int r = r0 + blockIdx.x, c = d - r;
  const int t = r * 32 + c;
  const int tid = threadIdx.x;
  const bool up_ok = t > 32;     // faithful off-by-one from the reference
  const bool left_ok = c > 0;

  // stage A = [h_up | h_left] (32 x 256) as bf16 in LDS
  for (int i = tid; i < BATCH * RR; i += 512) {
    int b = i >> 7, j = i & 127;
    float h1 = up_ok   ? H_all[(size_t)(t - 32) * BATCH * RR + i] : 0.0f;
    float h2 = left_ok ? H_all[(size_t)(t - 1)  * BATCH * RR + i] : 0.0f;
    As[b * AS_STRIDE + j]       = (__bf16)h1;
    As[b * AS_STRIDE + 128 + j] = (__bf16)h2;
  }
  __syncthreads();

  const int wave = tid >> 5, lane = tid & 31;
  const int mt = wave & 1, nb0 = (wave >> 1) * 5;
  const int M = mt * 16 + ((lane >> 4) & 1) * 8;
  const float* gx = gatesX + (size_t)t * BATCH * N5;
  f32x8 acc[5];
#pragma unroll
  for (int nt = 0; nt < 5; ++nt) {
    int N = (nb0 + nt) * 16 + (lane & 15);
    f32x8 cinit;
#pragma unroll
    for (int q = 0; q < 8; ++q) cinit[q] = gx[(size_t)(M + q) * N5 + N];
    acc[nt] = cinit;
  }
  // software pipeline: preload kb=0 operands
  bf16x16 bcur[5], bnxt[5];
#pragma unroll
  for (int nt = 0; nt < 5; ++nt) bcur[nt] = load_frag_b(whpack, nb0 + nt, lane);
  bf16x16 a = load_frag_a(As, AS_STRIDE, mt, 0, lane);
#pragma unroll
  for (int kb = 0; kb < 8; ++kb) {
    if (kb + 1 < 8) {
#pragma unroll
      for (int nt = 0; nt < 5; ++nt)
        bnxt[nt] = load_frag_b(whpack, (kb + 1) * NBT + nb0 + nt, lane);
    }
    bf16x16 anxt = (kb + 1 < 8) ? load_frag_a(As, AS_STRIDE, mt, kb + 1, lane) : a;
#pragma unroll
    for (int nt = 0; nt < 5; ++nt) acc[nt] = wmma_bf16(a, bcur[nt], acc[nt]);
    a = anxt;
#pragma unroll
    for (int nt = 0; nt < 5; ++nt) bcur[nt] = bnxt[nt];
  }
  // regroup gates through LDS
#pragma unroll
  for (int nt = 0; nt < 5; ++nt) {
    int N = (nb0 + nt) * 16 + (lane & 15);
#pragma unroll
    for (int q = 0; q < 8; ++q) gl[(M + q) * GL_STRIDE + N] = acc[nt][q];
  }
  __syncthreads();

  // elementwise LSTM cell
  for (int i = tid; i < BATCH * RR; i += 512) {
    int b = i >> 7, j = i & 127;
    const float* g = gl + b * GL_STRIDE;
    float gi = g[j], gj = g[128 + j], gf1 = g[256 + j], gf2 = g[384 + j],
          go = g[512 + j];
    float c1 = up_ok   ? C_all[(size_t)(t - 32) * BATCH * RR + i] : 0.0f;
    float c2 = left_ok ? C_all[(size_t)(t - 1)  * BATCH * RR + i] : 0.0f;
    float nc = c1 * sigmoidf_(gf1) + c2 * sigmoidf_(gf2) +
               sigmoidf_(gi) * tanhf(gj);
    float nh = tanhf(nc) * sigmoidf_(go);
    C_all[(size_t)t * BATCH * RR + i] = nc;
    H_all[(size_t)t * BATCH * RR + i] = nh;
    out[((size_t)b * 1024 + t) * RR + j] = nh;   // [B, h, w, R]
  }
}

// ---------------- host side ----------------
extern "C" void kernel_launch(void* const* d_in, const int* in_sizes, int n_in,
                              void* d_out, int out_size, void* d_ws,
                              size_t ws_size, hipStream_t stream) {
  const float* x    = (const float*)d_in[0];   // [32,32,128,128]
  const float* W    = (const float*)d_in[1];   // [768,640]
  const float* bias = (const float*)d_in[2];   // [640]
  float* out = (float*)d_out;
  char*  ws  = (char*)d_ws;

  __bf16* xs     = (__bf16*)(ws + OFF_XS);
  __bf16* wxpack = (__bf16*)(ws + OFF_WX);
  __bf16* whpack = (__bf16*)(ws + OFF_WH);
  float*  gatesX = (float*)(ws + OFF_GX);
  float*  C_all  = (float*)(ws + OFF_C);
  float*  H_all  = (float*)(ws + OFF_H);

  // 1) reorder/convert inputs to bf16
  mdlstm_pack_xs<<<65536, 256, 0, stream>>>(x, xs);
  // 2) pack weights into WMMA B-fragment order (bf16)
  mdlstm_pack_w<<<120, 256, 0, stream>>>(W, wxpack, whpack);
  // 3) hoisted input projection: one big parallel GEMM (+bias)
  mdlstm_gemm_x<<<1024, 512, 0, stream>>>(xs, wxpack, bias, gatesX);
  // 4) 63 wavefront anti-diagonals; stream order provides the dependency sync
  const size_t smem = (size_t)32 * AS_STRIDE * 2 + (size_t)32 * GL_STRIDE * 4;
  for (int d = 0; d < 63; ++d) {
    int cnt = (d < 32) ? (d + 1) : (63 - d);
    mdlstm_scan_diag<<<cnt, 512, smem, stream>>>(whpack, gatesX, C_all, H_all,
                                                 out, d);
  }
}
